// ALiBiAttention_4896262718017
// MI455X (gfx1250) — matrix operand order repaired and checked by reading, not compiled, not run on hardware
//
#include <hip/hip_runtime.h>
#include <hip/hip_bf16.h>
#include <stdint.h>

// ---------------------------------------------------------------------------
// ALiBi multi-head attention forward, MI455X (gfx1250, wave32, WMMA bf16).
// B=4, S=1024, D=1024, H=16, Dk=64.
// d_out = [out (4M fp32)] ++ [attn (64M fp32)]
// ---------------------------------------------------------------------------

#define B_   4
#define S_   1024
#define D_   1024
#define H_   16
#define DK_  64
#define BS_  (B_ * S_)          // 4096 rows for the big GEMMs

typedef __attribute__((ext_vector_type(16))) __bf16 v16bf;
typedef __attribute__((ext_vector_type(8)))  float  v8f;

union FragBF {
    v16bf v;
    uint4 q[2];
};

__device__ __forceinline__ unsigned short f32_to_bf16(float f) {
    unsigned int u = __float_as_uint(f);
    u += 0x7FFFu + ((u >> 16) & 1u);       // round-to-nearest-even
    return (unsigned short)(u >> 16);
}

// A operand (16xK tile, row-major K-contiguous source).
// Lanes 0-15: row m, K = kb+0..15; lanes 16-31: row m, K = kb+16..31.
__device__ __forceinline__ FragBF load_frag_a(const unsigned short* base, int ld,
                                              int m0, int kb, int lane) {
    FragBF f;
    const int m  = m0 + (lane & 15);
    const int k0 = kb + ((lane >> 4) << 4);
    const unsigned short* p = base + (size_t)m * ld + k0;
    f.q[0] = *reinterpret_cast<const uint4*>(p);
    f.q[1] = *reinterpret_cast<const uint4*>(p + 8);
    return f;
}

// B operand (Kx16 tile; source row n holds K contiguous).
// Lanes 0-15: col n, K = kb+0..15; lanes 16-31: col n, K = kb+16..31.
__device__ __forceinline__ FragBF load_frag_b(const unsigned short* base, int ld,
                                              int n0, int kb, int lane) {
    FragBF f;
    const int n  = n0 + (lane & 15);
    const int k0 = kb + ((lane >> 4) << 4);
    const unsigned short* p = base + (size_t)n * ld + k0;
    f.q[0] = *reinterpret_cast<const uint4*>(p);
    f.q[1] = *reinterpret_cast<const uint4*>(p + 8);
    return f;
}

__device__ __forceinline__ v8f wmma_bf16(const FragBF& a, const FragBF& b, v8f c) {
    return __builtin_amdgcn_wmma_f32_16x16x32_bf16(
        false, a.v, false, b.v, (short)0, c, false, false);
}

// ---------------------------------------------------------------------------
// fp32 -> bf16 conversion (one-time, feeds the WMMA pipeline)
// ---------------------------------------------------------------------------
__global__ __launch_bounds__(256) void cvt_f32_to_bf16(const float* __restrict__ src,
                                                       unsigned short* __restrict__ dst,
                                                       int n) {
    int i = blockIdx.x * blockDim.x + threadIdx.x;
    if (i < n) dst[i] = f32_to_bf16(src[i]);
}

// ---------------------------------------------------------------------------
// GEMM: y[m,n] = sum_k A[m,k] * W[n,k] + bias[n]   (M=4096, N=K=1024)
// mode 0: store bf16 as Q/K layout  [B,H,S,Dk]
// mode 2: store bf16 as Vt layout   [B,H,Dk,S]
// mode 3: store fp32 row-major to d_out (output projection)
//
// Block: 128 threads (4 waves). Block tile 128x64; each wave 32x64 (8 WMMA
// accumulators). W tile (64x32 bf16) is double-buffered in LDS via
// global_load_async_to_lds_b128 (ASYNCcnt) so all 4 waves share one copy.
// A fragments are register-prefetched one k-step ahead.
// ---------------------------------------------------------------------------
#define WT_LDR 40   // LDS row stride in ushorts (80B: 16B-aligned, conflict-skewed)

__global__ __launch_bounds__(128) void gemm_bf16(const unsigned short* __restrict__ A,
                                                 const unsigned short* __restrict__ W,
                                                 const float* __restrict__ bias,
                                                 unsigned short* __restrict__ dst_bf,
                                                 float* __restrict__ dst_f,
                                                 int mode) {
    __shared__ __align__(16) unsigned short wt[2][64][WT_LDR];  // 10 KB

    const int tid  = threadIdx.x;
    const int lane = tid & 31;
    const int wv   = tid >> 5;
    const int m0   = blockIdx.y * 128 + wv * 32;
    const int n0   = blockIdx.x * 64;

    // async-stage one 64x32 bf16 W tile into LDS buffer s (2 b128 per thread)
    auto issue_copy = [&](int kb, int s) {
#pragma unroll
        for (int jj = 0; jj < 2; ++jj) {
            const int j   = tid + jj * 128;       // 256 x b128 transfers total
            const int row = j >> 2;
            const int seg = j & 3;
            const unsigned short* g = W + (size_t)(n0 + row) * D_ + kb + seg * 8;
            const unsigned lds = (unsigned)(size_t)&wt[s][row][seg * 8];
            asm volatile("global_load_async_to_lds_b128 %0, %1, off"
                         :: "v"(lds), "v"((unsigned long long)(size_t)g)
                         : "memory");
        }
    };

    v8f acc[2][4] = {};
    issue_copy(0, 0);
    int s = 0;

    FragBF a0 = load_frag_a(A, D_, m0,      0, lane);
    FragBF a1 = load_frag_a(A, D_, m0 + 16, 0, lane);

    for (int kb = 0; kb < D_; kb += 32) {
        const bool more = (kb + 32) < D_;
        if (more) issue_copy(kb + 32, s ^ 1);           // prefetch next W tile
        if (more) asm volatile("s_wait_asynccnt 0x2" ::: "memory"); // stage s done
        else      asm volatile("s_wait_asynccnt 0x0" ::: "memory");
        __syncthreads();

        FragBF na0, na1;                                 // prefetch next A frags
        if (more) {
            na0 = load_frag_a(A, D_, m0,      kb + 32, lane);
            na1 = load_frag_a(A, D_, m0 + 16, kb + 32, lane);
        }

#pragma unroll
        for (int t = 0; t < 4; ++t) {
            FragBF b;
            const unsigned short* p = &wt[s][t * 16 + (lane & 15)][(lane >> 4) << 4];
            b.q[0] = *reinterpret_cast<const uint4*>(p);
            b.q[1] = *reinterpret_cast<const uint4*>(p + 8);
            acc[0][t] = wmma_bf16(a0, b, acc[0][t]);
            acc[1][t] = wmma_bf16(a1, b, acc[1][t]);
        }
        __syncthreads();    // everyone done with buffer s before it is re-filled
        if (more) { a0 = na0; a1 = na1; }
        s ^= 1;
    }

    const int mrow = (lane >> 4) << 3;   // 0 or 8
    const int ncol = lane & 15;
#pragma unroll
    for (int r2 = 0; r2 < 2; ++r2) {
#pragma unroll
        for (int t = 0; t < 4; ++t) {
            const int   gn = n0 + t * 16 + ncol;
            const float bv = bias[gn];
#pragma unroll
            for (int r = 0; r < 8; ++r) {
                const int   gm  = m0 + r2 * 16 + mrow + r;
                const float val = acc[r2][t][r] + bv;
                if (mode == 3) {
                    dst_f[(size_t)gm * D_ + gn] = val;
                } else {
                    const int bb = gm >> 10, ss = gm & (S_ - 1);
                    const int hh = gn >> 6,  d  = gn & (DK_ - 1);
                    size_t idx;
                    if (mode == 2)  // V transposed: [B,H,Dk,S]
                        idx = (((size_t)(bb * H_ + hh)) * DK_ + d) * S_ + ss;
                    else            // Q/K: [B,H,S,Dk]
                        idx = (((size_t)(bb * H_ + hh)) * S_ + ss) * DK_ + d;
                    dst_bf[idx] = f32_to_bf16(val);
                }
            }
        }
    }
}

// ---------------------------------------------------------------------------
// Fused attention: per (b, h, 16-query block):
//   scores (WMMA, scale + ALiBi) -> LDS -> softmax -> attn NT-store + bf16 in
//   LDS (in-place repack) -> PV (WMMA) -> ctx bf16 [B*S, D]
// Block: 128 threads (4 waves). LDS: 16x1024 fp32 = 64 KB.
// ---------------------------------------------------------------------------
__global__ __launch_bounds__(128) void attn_fused(const unsigned short* __restrict__ Q,
                                                  const unsigned short* __restrict__ K,
                                                  const unsigned short* __restrict__ Vt,
                                                  float* __restrict__ attn_out,
                                                  unsigned short* __restrict__ ctx_bf) {
    __shared__ float sc[16 * S_];   // 64 KB

    const int lane = threadIdx.x & 31;
    const int wv   = threadIdx.x >> 5;
    const int b    = blockIdx.z;
    const int h    = blockIdx.y;
    const int qb   = blockIdx.x << 4;

    const size_t headoff = ((size_t)(b * H_ + h)) * S_ * DK_;
    const unsigned short* Qp = Q  + headoff;
    const unsigned short* Kp = K  + headoff;
    const unsigned short* Vp = Vt + headoff;

    const float slope = exp2f(-0.5f * (float)(h + 1));  // 2^(-8(h+1)/16)
    const int   mrow  = (lane >> 4) << 3;

    // ---- Phase 1: scores = Q K^T / 8 + alibi -> LDS -------------------------
    FragBF aq0 = load_frag_a(Qp, DK_, qb, 0,  lane);
    FragBF aq1 = load_frag_a(Qp, DK_, qb, 32, lane);

    for (int t = 0; t < 16; ++t) {            // each wave: 16 key tiles
        const int col0 = ((wv * 16) + t) << 4;
        v8f acc = {};
        FragBF bk0 = load_frag_b(Kp, DK_, col0, 0,  lane);
        acc = wmma_bf16(aq0, bk0, acc);
        FragBF bk1 = load_frag_b(Kp, DK_, col0, 32, lane);
        acc = wmma_bf16(aq1, bk1, acc);

        const int kcol = col0 + (lane & 15);
#pragma unroll
        for (int r = 0; r < 8; ++r) {
            const int   m    = mrow + r;
            const float qpos = (float)(qb + m);
            sc[m * S_ + kcol] = acc[r] * 0.125f - slope * fabsf(qpos - (float)kcol);
        }
    }
    __syncthreads();

    // ---- Phase 2: row softmax; stream attn to HBM; repack bf16 in place ----
    for (int rr = 0; rr < 4; ++rr) {
        const int row  = wv + rr * 4;
        float*    srow = &sc[row * S_];

        float vmax = -1e30f;
        for (int i = 0; i < 32; ++i) vmax = fmaxf(vmax, srow[i * 32 + lane]);
#pragma unroll
        for (int off = 16; off >= 1; off >>= 1)
            vmax = fmaxf(vmax, __shfl_xor(vmax, off, 32));

        float lsum = 0.f;
        for (int i = 0; i < 32; ++i) {
            const int k = i * 32 + lane;
            const float e = __expf(srow[k] - vmax);
            srow[k] = e;
            lsum += e;
        }
#pragma unroll
        for (int off = 16; off >= 1; off >>= 1)
            lsum += __shfl_xor(lsum, off, 32);
        const float inv = 1.0f / lsum;

        float* gout = attn_out +
            ((((size_t)(b * H_ + h)) * S_) + (size_t)(qb + row)) * S_;
        unsigned short* arow = reinterpret_cast<unsigned short*>(srow);
        for (int i = 0; i < 32; ++i) {
            const int   k = i * 32 + lane;
            const float a = srow[k] * inv;
            __builtin_nontemporal_store(a, &gout[k]);   // attn never re-read from HBM
            arow[k] = f32_to_bf16(a);   // safe: wave-lockstep, reads precede writes
        }
    }
    __syncthreads();

    // ---- Phase 3: ctx = attn @ V  (A from LDS bf16, B from Vt) -------------
    const unsigned short* attbase = reinterpret_cast<const unsigned short*>(sc);
    v8f acc = {};
    for (int kb = 0; kb < S_; kb += 32) {
        FragBF a = load_frag_a(attbase, 2 * S_, 0, kb, lane); // row stride 2048 ushorts
        FragBF v = load_frag_b(Vp, S_, wv * 16, kb, lane);
        acc = wmma_bf16(a, v, acc);
    }

    const int ncol = lane & 15;
#pragma unroll
    for (int r = 0; r < 8; ++r) {
        const int m    = mrow + r;
        const int grow = b * S_ + qb + m;
        const int gcol = h * DK_ + wv * 16 + ncol;
        ctx_bf[(size_t)grow * D_ + gcol] = f32_to_bf16(acc[r]);
    }
}

// ---------------------------------------------------------------------------
// Host-side launch
// ---------------------------------------------------------------------------
extern "C" void kernel_launch(void* const* d_in, const int* in_sizes, int n_in,
                              void* d_out, int out_size, void* d_ws, size_t ws_size,
                              hipStream_t stream) {
    const float* x  = (const float*)d_in[0];
    const float* Wq = (const float*)d_in[1];
    const float* bq = (const float*)d_in[2];
    const float* Wk = (const float*)d_in[3];
    const float* bk = (const float*)d_in[4];
    const float* Wv = (const float*)d_in[5];
    const float* bv = (const float*)d_in[6];
    const float* Wo = (const float*)d_in[7];
    const float* bo = (const float*)d_in[8];

    float* out      = (float*)d_out;                     // [4096, 1024]
    float* attn_out = out + (size_t)BS_ * D_;            // [B,H,S,S]

    // workspace layout (bytes)
    char* ws = (char*)d_ws;
    const size_t MB = 1u << 20;
    unsigned short* xbf  = (unsigned short*)(ws +  0 * MB);  // 8 MB  [4096,1024]
    unsigned short* wqbf = (unsigned short*)(ws +  8 * MB);  // 2 MB
    unsigned short* wkbf = (unsigned short*)(ws + 10 * MB);  // 2 MB
    unsigned short* wvbf = (unsigned short*)(ws + 12 * MB);  // 2 MB
    unsigned short* wobf = (unsigned short*)(ws + 14 * MB);  // 2 MB
    unsigned short* qbf  = (unsigned short*)(ws + 16 * MB);  // 8 MB  [B,H,S,Dk]
    unsigned short* kbf  = (unsigned short*)(ws + 24 * MB);  // 8 MB  [B,H,S,Dk]
    unsigned short* vtbf = (unsigned short*)(ws + 32 * MB);  // 8 MB  [B,H,Dk,S]
    unsigned short* ctxb = (unsigned short*)(ws + 40 * MB);  // 8 MB  [4096,1024]

    const int nx = BS_ * D_;      // 4,194,304
    const int nw = D_ * D_;       // 1,048,576

    cvt_f32_to_bf16<<<nx / 256, 256, 0, stream>>>(x,  xbf,  nx);
    cvt_f32_to_bf16<<<nw / 256, 256, 0, stream>>>(Wq, wqbf, nw);
    cvt_f32_to_bf16<<<nw / 256, 256, 0, stream>>>(Wk, wkbf, nw);
    cvt_f32_to_bf16<<<nw / 256, 256, 0, stream>>>(Wv, wvbf, nw);
    cvt_f32_to_bf16<<<nw / 256, 256, 0, stream>>>(Wo, wobf, nw);

    dim3 ggrid(D_ / 64, BS_ / 128, 1);   // (16, 32): block tile 128x64
    gemm_bf16<<<ggrid, 128, 0, stream>>>(xbf, wqbf, bq, qbf,  nullptr, 0);
    gemm_bf16<<<ggrid, 128, 0, stream>>>(xbf, wkbf, bk, kbf,  nullptr, 0);
    gemm_bf16<<<ggrid, 128, 0, stream>>>(xbf, wvbf, bv, vtbf, nullptr, 2);

    dim3 agrid(S_ / 16, H_, B_);        // (64, 16, 4)
    attn_fused<<<agrid, 128, 0, stream>>>(qbf, kbf, vtbf, attn_out, ctxb);

    gemm_bf16<<<ggrid, 128, 0, stream>>>(ctxb, wobf, bo, nullptr, out, 3);
}
